// GPTNet_63136019251508
// MI455X (gfx1250) — matrix operand order repaired and checked by reading, not compiled, not run on hardware
//
#include <hip/hip_runtime.h>
#include <math.h>

// ---------------------------------------------------------------------------
// GPTNet block for MI455X (gfx1250): wave32 + V_WMMA_F32_16X16X32_BF16
// - branch-free vectorized (b128) fragment loads
// - pre-packed bf16 weights (B operand = one 32B vector load per tile)
// - in-projection GEMM scatters q/k directly (no repack pass)
// - attention-apply output stored as bf16 WMMA-A fragments (no cvt in out-proj)
// ---------------------------------------------------------------------------

typedef __bf16 bf16_t;
typedef bf16_t v16bf __attribute__((ext_vector_type(16)));
typedef float  v8f   __attribute__((ext_vector_type(8)));

union ABu { v16bf v; bf16_t e[16]; };
union DDu { v8f   v; float  e[8];  };

// problem dims
constexpr int N_  = 128;
constexpr int T_  = 32;
constexpr int V_  = 25;
constexpr int C_  = 256;
constexpr int IC_ = 64;
constexpr int S_  = 3;
constexpr int ROWS = N_ * T_ * V_;   // 102400
constexpr int QKC  = 2 * S_ * IC_;   // 384
constexpr int SC   = S_ * C_;        // 768
constexpr int ATT_STRIDE = 1024;     // per-(n,s) slot: padded 32x32, ld = 32
constexpr int ATT_LD = 32;

// ---------------------------------------------------------------------------
// WMMA + fragment helpers
// ---------------------------------------------------------------------------
__device__ inline v8f wmma_bf16(v16bf a, v16bf b, v8f c) {
  return __builtin_amdgcn_wmma_f32_16x16x32_bf16(false, a, false, b, (short)0, c,
                                                 false, false);
}

__device__ inline v16bf cvt16(float4 f0, float4 f1, float4 f2, float4 f3) {
  ABu a;
  a.e[0]  = (bf16_t)f0.x; a.e[1]  = (bf16_t)f0.y; a.e[2]  = (bf16_t)f0.z; a.e[3]  = (bf16_t)f0.w;
  a.e[4]  = (bf16_t)f1.x; a.e[5]  = (bf16_t)f1.y; a.e[6]  = (bf16_t)f1.z; a.e[7]  = (bf16_t)f1.w;
  a.e[8]  = (bf16_t)f2.x; a.e[9]  = (bf16_t)f2.y; a.e[10] = (bf16_t)f2.z; a.e[11] = (bf16_t)f2.w;
  a.e[12] = (bf16_t)f3.x; a.e[13] = (bf16_t)f3.y; a.e[14] = (bf16_t)f3.z; a.e[15] = (bf16_t)f3.w;
  return a.v;
}

__device__ inline v16bf load_8x2(const float* __restrict__ g0,
                                 const float* __restrict__ g1) {
  const float4* p0 = (const float4*)g0;
  const float4* p1 = (const float4*)g1;
  return cvt16(p0[0], p0[1], p1[0], p1[1]);
}

// A fragment (16x32 tile, row-major f32): `Arow` already includes (lane&15) row.
__device__ inline v16bf frag_a(const float* __restrict__ Arow, int k0, int hi) {
  const float* g = Arow + k0 + (hi ? 16 : 0);
  return load_8x2(g, g + 8);
}

// B^T-style fragment: per-lane 16 contiguous k at rowbase + k0 + (hi?16:0)
__device__ inline v16bf frag_bT(const float* __restrict__ Brow, int k0, int hi) {
  const float* g = Brow + k0 + (hi ? 16 : 0);
  return load_8x2(g, g + 8);
}

// bf16 element position inside the packed A-fragment stream for (row m, col k)
__device__ inline size_t apack_idx(int m, int k, int ktc /*K/32*/) {
  int kk = k & 31;
  int hi = kk >> 4;
  int e = kk & 15;
  int lane = (m & 15) + (hi << 4);
  return ((((size_t)(m >> 4) * ktc + (k >> 5)) << 5) + lane) * 16 + e;
}

__device__ inline float wave_sum(float v) {
#pragma unroll
  for (int off = 16; off > 0; off >>= 1) v += __shfl_xor(v, off, 32);
  return v;
}

__device__ inline float gelu_exact(float g) {
  return 0.5f * g * (1.0f + erff(g * 0.70710678118654752f));
}

// ---------------------------------------------------------------------------
// Weight packing: W (K x Nc, f32) -> bf16 B fragments.
// Bp[((kt*ntN)+nt)*32 + lane].e[i] = W[kt*32+(hi?16:0)+i][nt*16+lane15]
// ---------------------------------------------------------------------------
__global__ void k_pack_b(const float* __restrict__ W, v16bf* __restrict__ Bp,
                         int K, int Nc) {
  int idx = blockIdx.x * blockDim.x + threadIdx.x;
  int ntN = Nc >> 4;
  int total = (K >> 5) * ntN * 32;
  if (idx >= total) return;
  int lane = idx & 31, hi = lane >> 4;
  int tile = idx >> 5;
  int kt = tile / ntN, nt = tile % ntN;
  int nc = nt * 16 + (lane & 15);
  int kb = kt * 32 + (hi ? 16 : 0);
  ABu b;
#pragma unroll
  for (int i = 0; i < 16; ++i) b.e[i] = (bf16_t)W[(size_t)(kb + i) * Nc + nc];
  Bp[idx] = b.v;
}

// ---------------------------------------------------------------------------
// Dense GEMM, f32 A: one wave = 16x64 strip (4 WMMA tiles), packed-bf16 B.
// mode 0: Cm[m*Nc+nc] = acc + bias                     (plain, no activation)
// mode 1: spatial q/k scatter with elu+1               (Nc == 384)
// mode 2: temporal q/k scatter with elu+1              (Nc == 384)
// ---------------------------------------------------------------------------
__global__ void k_gemm(const float* __restrict__ A, const v16bf* __restrict__ Bp,
                       const float* __restrict__ bias, float* __restrict__ Cm,
                       float* __restrict__ Qr, float* __restrict__ Kr,
                       int M, int K, int Nc, int mode) {
  int wid = blockIdx.x * (blockDim.x >> 5) + (threadIdx.x >> 5);
  int nChunks = Nc >> 6;
  if (wid >= (M >> 4) * nChunks) return;  // wave-uniform
  int tm = wid / nChunks, tc = wid % nChunks;
  int lane = threadIdx.x & 31, hi = lane >> 4, l15 = lane & 15;
  int ntN = Nc >> 4;

  DDu acc[4];
#pragma unroll
  for (int q = 0; q < 4; ++q)
#pragma unroll
    for (int i = 0; i < 8; ++i) acc[q].e[i] = 0.0f;

  const float* Arow = A + (size_t)(tm * 16 + l15) * K;
  int ktc = K >> 5;
  for (int kt = 0; kt < ktc; ++kt) {
    v16bf a = frag_a(Arow, kt * 32, hi);
    const v16bf* bb = Bp + ((size_t)kt * ntN + tc * 4) * 32 + lane;
#pragma unroll
    for (int q = 0; q < 4; ++q) {
      v16bf b = bb[q * 32];
      acc[q].v = wmma_bf16(a, b, acc[q].v);
    }
  }

#pragma unroll
  for (int q = 0; q < 4; ++q) {
    int nc = (tc * 4 + q) * 16 + l15;
    float bv = bias[nc];
    if (mode == 0) {
#pragma unroll
      for (int r = 0; r < 8; ++r) {
        int m = tm * 16 + r + (hi ? 8 : 0);
        Cm[(size_t)m * Nc + nc] = acc[q].e[r] + bv;
      }
    } else {
      // column nc = ic*2S + ss  (qk reshape);  ss<S -> q, else k
      int ic = nc / (2 * S_), ss = nc % (2 * S_);
      int s = (ss < S_) ? ss : ss - S_;
      float* dst = (ss < S_) ? Qr : Kr;
#pragma unroll
      for (int r = 0; r < 8; ++r) {
        int m = tm * 16 + r + (hi ? 8 : 0);  // m = (n*T + t)*V + v
        int n = m / (T_ * V_);
        int t = (m / V_) % T_;
        int v = m % V_;
        float val = acc[q].e[r] + bv;
        val = (val > 0.0f) ? (val + 1.0f) : expf(val);  // elu+1
        size_t idx = (mode == 1)
            ? ((size_t)((n * S_ + s) * V_ + v)) * (T_ * IC_) + (size_t)t * IC_ + ic
            : ((size_t)((n * S_ + s) * T_ + t)) * (V_ * IC_) + (size_t)v * IC_ + ic;
        dst[idx] = val;
      }
    }
  }
}

// ---------------------------------------------------------------------------
// Dense GEMM, packed-bf16 A (out-projections): zero conversions in hot loop.
// One wave = 16x64 strip; A = one v16bf per lane per k-step.
// ---------------------------------------------------------------------------
__global__ void k_gemm_paA(const v16bf* __restrict__ Ap, const v16bf* __restrict__ Bp,
                           const float* __restrict__ bias, float* __restrict__ Cm,
                           int M, int K, int Nc) {
  int wid = blockIdx.x * (blockDim.x >> 5) + (threadIdx.x >> 5);
  int nChunks = Nc >> 6;
  if (wid >= (M >> 4) * nChunks) return;
  int tm = wid / nChunks, tc = wid % nChunks;
  int lane = threadIdx.x & 31, hi = lane >> 4, l15 = lane & 15;
  int ntN = Nc >> 4;
  int ktc = K >> 5;

  DDu acc[4];
#pragma unroll
  for (int q = 0; q < 4; ++q)
#pragma unroll
    for (int i = 0; i < 8; ++i) acc[q].e[i] = 0.0f;

  const v16bf* Abase = Ap + ((size_t)tm * ktc << 5) + lane;
  for (int kt = 0; kt < ktc; ++kt) {
    v16bf a = Abase[kt << 5];
    const v16bf* bb = Bp + ((size_t)kt * ntN + tc * 4) * 32 + lane;
#pragma unroll
    for (int q = 0; q < 4; ++q) {
      v16bf b = bb[q * 32];
      acc[q].v = wmma_bf16(a, b, acc[q].v);
    }
  }

#pragma unroll
  for (int q = 0; q < 4; ++q) {
    int nc = (tc * 4 + q) * 16 + l15;
    float bv = bias[nc];
#pragma unroll
    for (int r = 0; r < 8; ++r) {
      int m = tm * 16 + r + (hi ? 8 : 0);
      Cm[(size_t)m * Nc + nc] = acc[q].e[r] + bv;
    }
  }
}

// ---------------------------------------------------------------------------
// Batched Att = Q @ K^T. Per batch: Mrows x Kdim row-major (Kdim%32==0).
// Stores FULL padded 32x32 tile (ld=32); norm kernels zero the pad columns.
// Q/K buffers carry slack rows so unchecked pad-row loads stay in-bounds.
// ---------------------------------------------------------------------------
__global__ void k_qkT(const float* __restrict__ Q, const float* __restrict__ Kd,
                      float* __restrict__ Att, int Mrows, int Kdim) {
  int batch = blockIdx.y;
  int w = threadIdx.x >> 5;
  int tm = w >> 1, tn = w & 1;
  int lane = threadIdx.x & 31, hi = lane >> 4, l15 = lane & 15;

  const float* Arow = Q + (size_t)batch * Mrows * Kdim + (size_t)(tm * 16 + l15) * Kdim;
  const float* Brow = Kd + (size_t)batch * Mrows * Kdim + (size_t)(tn * 16 + l15) * Kdim;

  DDu acc;
#pragma unroll
  for (int i = 0; i < 8; ++i) acc.e[i] = 0.0f;

  for (int k0 = 0; k0 < Kdim; k0 += 32) {
    v16bf a = frag_a(Arow, k0, hi);
    v16bf b = frag_bT(Brow, k0, hi);
    acc.v = wmma_bf16(a, b, acc.v);
  }

  float* Ab = Att + (size_t)batch * ATT_STRIDE;
  int nc = tn * 16 + l15;
#pragma unroll
  for (int r = 0; r < 8; ++r) {
    int m = tm * 16 + r + (hi ? 8 : 0);
    Ab[m * ATT_LD + nc] = acc.e[r];
  }
}

// ---------------------------------------------------------------------------
// Attention row-normalization (1/(IC*T) scale cancels under row-sum division).
// ---------------------------------------------------------------------------
__global__ void k_att_norm_spatial(float* __restrict__ Att,
                                   const float* __restrict__ att0) {
  int row = blockIdx.x * (blockDim.x >> 5) + (threadIdx.x >> 5);  // (n,s,v)
  if (row >= N_ * S_ * V_) return;
  int lane = threadIdx.x & 31;
  int v = row % V_;
  int s = (row / V_) % S_;
  int batch = row / V_;  // n*S + s
  float* rp = Att + (size_t)batch * ATT_STRIDE + (size_t)v * ATT_LD;
  float val = (lane < V_) ? rp[lane] : 0.0f;
  float sum = wave_sum(val);
  float res = (lane < V_) ? val / sum + att0[((size_t)s * V_ + v) * V_ + lane] : 0.0f;
  rp[lane] = res;  // zero pad columns -> kills garbage/NaN in k=25..31
}

__global__ void k_att_norm_temporal(float* __restrict__ Att, int upper) {
  int row = blockIdx.x * (blockDim.x >> 5) + (threadIdx.x >> 5);  // (n,s,t)
  if (row >= N_ * S_ * T_) return;
  int lane = threadIdx.x & 31;
  int t = row % T_;
  int batch = row / T_;
  float* rp = Att + (size_t)batch * ATT_STRIDE + (size_t)t * ATT_LD;
  float val = rp[lane];
  bool keep = upper ? (lane >= t) : (lane <= t);
  val = keep ? val : 0.0f;
  float sum = wave_sum(val);
  rp[lane] = val / sum;
}

// ---------------------------------------------------------------------------
// Spatial apply: YP(bf16, A-fragment order)[(n,t,v), c*S+s] =
//   sum_u att[n,s,v,u] * x[n,t,u,c]
// A = padded 32x32 att (cols>=25 exact zeros); x rows u>=25 address-clamped.
// ---------------------------------------------------------------------------
__global__ void k_spatial_apply(const float* __restrict__ Att,
                                const float* __restrict__ x,
                                bf16_t* __restrict__ YP) {
  int batch = blockIdx.y;
  int n = batch / S_, s = batch % S_;
  int w = blockIdx.x * 4 + (threadIdx.x >> 5);
  int tn = w >> 1, tm = w & 1;
  int lane = threadIdx.x & 31, hi = lane >> 4, l15 = lane & 15;

  const float* Arow = Att + (size_t)batch * ATT_STRIDE + (size_t)(tm * 16 + l15) * ATT_LD;
  v16bf a = frag_a(Arow, 0, hi);

  int nc = tn * 16 + l15;  // (t,c) flat in [0, 8192)
  int t = nc >> 8, c = nc & 255;
  const float* xb = x + ((size_t)(n * T_ + t) * V_) * C_ + c;
  ABu b;
#pragma unroll
  for (int j = 0; j < 8; ++j) {
#pragma unroll
    for (int h = 0; h < 2; ++h) {
      int u = (hi ? 16 : 0) + 2 * j + h;
      int uc = (u < V_) ? u : (V_ - 1);  // clamp: unconditional in-bounds load
      b.e[2 * j + h] = (bf16_t)xb[(size_t)uc * C_];
    }
  }

  DDu acc;
#pragma unroll
  for (int i = 0; i < 8; ++i) acc.e[i] = 0.0f;
  acc.v = wmma_bf16(a, b.v, acc.v);

  int col = c * S_ + s;
#pragma unroll
  for (int r = 0; r < 8; ++r) {
    int v = tm * 16 + r + (hi ? 8 : 0);
    if (v < V_) {
      int m = (n * T_ + t) * V_ + v;
      YP[apack_idx(m, col, SC / 32)] = (bf16_t)acc.e[r];
    }
  }
}

// ---------------------------------------------------------------------------
// Temporal apply: YP(bf16, A-fragment order)[(n,t,v), c*S+s] =
//   sum_k a[n,s,t,k] * y[n,k,v,c]
// ---------------------------------------------------------------------------
__global__ void k_temporal_apply(const float* __restrict__ Att,
                                 const float* __restrict__ Y,
                                 bf16_t* __restrict__ YP) {
  int batch = blockIdx.y;
  int n = batch / S_, s = batch % S_;
  int w = blockIdx.x * 4 + (threadIdx.x >> 5);
  int tn = w >> 1, tm = w & 1;
  int lane = threadIdx.x & 31, hi = lane >> 4, l15 = lane & 15;

  const float* Arow = Att + (size_t)batch * ATT_STRIDE + (size_t)(tm * 16 + l15) * ATT_LD;
  v16bf a = frag_a(Arow, 0, hi);

  int nc = tn * 16 + l15;  // (v,c) flat in [0, 6400)
  const float* yb = Y + (size_t)n * T_ * V_ * C_ + nc;
  ABu b;
#pragma unroll
  for (int j = 0; j < 8; ++j) {
#pragma unroll
    for (int h = 0; h < 2; ++h) {
      int k = (hi ? 16 : 0) + 2 * j + h;
      b.e[2 * j + h] = (bf16_t)yb[(size_t)k * (V_ * C_)];
    }
  }

  DDu acc;
#pragma unroll
  for (int i = 0; i < 8; ++i) acc.e[i] = 0.0f;
  acc.v = wmma_bf16(a, b.v, acc.v);

  int v = nc / C_, c = nc % C_;
  int col = c * S_ + s;
#pragma unroll
  for (int r = 0; r < 8; ++r) {
    int t = tm * 16 + r + (hi ? 8 : 0);
    int m = (n * T_ + t) * V_ + v;
    YP[apack_idx(m, col, SC / 32)] = (bf16_t)acc.e[r];
  }
}

// ---------------------------------------------------------------------------
// LayerNorm over 256 cols, optionally + residual then exact GELU.
// ---------------------------------------------------------------------------
__global__ void k_ln_res_gelu(const float* __restrict__ inp,
                              const float* __restrict__ resid,
                              float* __restrict__ out, int rows) {
  int row = blockIdx.x * (blockDim.x >> 5) + (threadIdx.x >> 5);
  if (row >= rows) return;
  int lane = threadIdx.x & 31;
  const float* ip = inp + (size_t)row * C_;
  float vals[8];
  float sum = 0.0f, sq = 0.0f;
#pragma unroll
  for (int j = 0; j < 8; ++j) {
    float v = ip[lane + 32 * j];
    vals[j] = v;
    sum += v;
    sq += v * v;
  }
  sum = wave_sum(sum);
  sq = wave_sum(sq);
  float mean = sum * (1.0f / C_);
  float var = sq * (1.0f / C_) - mean * mean;
  float rstd = rsqrtf(var + 1e-5f);
  float* op = out + (size_t)row * C_;
  if (resid) {
    const float* rp = resid + (size_t)row * C_;
#pragma unroll
    for (int j = 0; j < 8; ++j) {
      float g = rp[lane + 32 * j] + (vals[j] - mean) * rstd;
      op[lane + 32 * j] = gelu_exact(g);
    }
  } else {
#pragma unroll
    for (int j = 0; j < 8; ++j) op[lane + 32 * j] = (vals[j] - mean) * rstd;
  }
}

__global__ void k_add3_gelu(const float* __restrict__ a, const float* __restrict__ b,
                            const float* __restrict__ c, float* __restrict__ out,
                            long nelem) {
  long i = (long)blockIdx.x * blockDim.x + threadIdx.x;
  if (i >= nelem) return;
  out[i] = gelu_exact(a[i] + b[i] + c[i]);
}

// ---------------------------------------------------------------------------
// Host orchestration
// ---------------------------------------------------------------------------
extern "C" void kernel_launch(void* const* d_in, const int* in_sizes, int n_in,
                              void* d_out, int out_size, void* d_ws, size_t ws_size,
                              hipStream_t stream) {
  const float* x    = (const float*)d_in[0];
  const float* inW  = (const float*)d_in[1];
  const float* inB  = (const float*)d_in[2];
  const float* att0 = (const float*)d_in[3];
  const float* outW = (const float*)d_in[4];
  const float* outB = (const float*)d_in[5];
  const float* ffW  = (const float*)d_in[6];
  const float* ffB  = (const float*)d_in[7];
  const float* tfW  = (const float*)d_in[8];
  const float* tfB  = (const float*)d_in[9];
  const float* tbW  = (const float*)d_in[10];
  const float* tbB  = (const float*)d_in[11];
  const float* ofW  = (const float*)d_in[12];
  const float* ofB  = (const float*)d_in[13];
  const float* obW  = (const float*)d_in[14];
  const float* obB  = (const float*)d_in[15];
  const float* ftW  = (const float*)d_in[16];
  const float* ftB  = (const float*)d_in[17];
  float* out = (float*)d_out;
  (void)ws_size; (void)in_sizes; (void)n_in; (void)out_size;

  float* ws = (float*)d_ws;
  size_t o = 0;
  const size_t QR_SLACK = 16384;  // pad-row slack for unchecked spatial loads
  float* QR  = ws + o; o += (size_t)N_ * S_ * V_ * T_ * IC_ + QR_SLACK;  // 19.7M
  float* KR  = ws + o; o += (size_t)N_ * S_ * V_ * T_ * IC_ + QR_SLACK;  // 19.7M
  float* ATT = ws + o; o += (size_t)N_ * S_ * ATT_STRIDE;                // 0.4M
  bf16_t* YP = (bf16_t*)(ws + o); o += (size_t)ROWS * SC / 2;            // 39.3M f32-equiv
  float* T1  = ws + o; o += (size_t)ROWS * C_;                           // 26.2M
  float* Yb  = ws + o; o += (size_t)ROWS * C_;
  float* ZF  = ws + o; o += (size_t)ROWS * C_;
  float* ZB  = ws + o; o += (size_t)ROWS * C_;
  float* Zb  = ws + o; o += (size_t)ROWS * C_;

  // packed bf16 weight region (32B fragments, offsets stay 64B aligned)
  v16bf* bp = (v16bf*)(ws + o);
  size_t bo = 0;
  auto carve = [&](int K, int Nc) { v16bf* p = bp + bo; bo += (size_t)(K / 32) * (Nc / 16) * 32; return p; };
  v16bf* P_inW = carve(C_, QKC);
  v16bf* P_outW = carve(SC, C_);
  v16bf* P_ffW = carve(C_, C_);
  v16bf* P_tfW = carve(C_, QKC);
  v16bf* P_tbW = carve(C_, QKC);
  v16bf* P_ofW = carve(SC, C_);
  v16bf* P_obW = carve(SC, C_);
  v16bf* P_ftW = carve(C_, C_);

  auto pack = [&](const float* W, v16bf* P, int K, int Nc) {
    int total = (K / 32) * (Nc / 16) * 32;
    k_pack_b<<<(total + 255) / 256, 256, 0, stream>>>(W, P, K, Nc);
  };
  pack(inW, P_inW, C_, QKC);
  pack(outW, P_outW, SC, C_);
  pack(ffW, P_ffW, C_, C_);
  pack(tfW, P_tfW, C_, QKC);
  pack(tbW, P_tbW, C_, QKC);
  pack(ofW, P_ofW, SC, C_);
  pack(obW, P_obW, SC, C_);
  pack(ftW, P_ftW, C_, C_);

  auto gemm = [&](const float* A, const v16bf* Bp, const float* bias, float* Cm,
                  float* Qr, float* Kr, int M, int K, int Nc, int mode) {
    int tiles = (M / 16) * (Nc / 64);
    k_gemm<<<(tiles + 3) / 4, 128, 0, stream>>>(A, Bp, bias, Cm, Qr, Kr, M, K, Nc, mode);
  };
  auto gemm_pa = [&](const bf16_t* Ap, const v16bf* Bp, const float* bias, float* Cm,
                     int M, int K, int Nc) {
    int tiles = (M / 16) * (Nc / 64);
    k_gemm_paA<<<(tiles + 3) / 4, 128, 0, stream>>>((const v16bf*)Ap, Bp, bias, Cm, M, K, Nc);
  };

  const int NB = N_ * S_;            // 384 attention batches
  const int LNB = (ROWS + 7) / 8;

  // ---- spatial block ----
  gemm(x, P_inW, inB, nullptr, QR, KR, ROWS, C_, QKC, 1);
  k_qkT<<<dim3(1, NB), 128, 0, stream>>>(QR, KR, ATT, V_, T_ * IC_);
  k_att_norm_spatial<<<(N_ * S_ * V_ + 3) / 4, 128, 0, stream>>>(ATT, att0);
  k_spatial_apply<<<dim3(256, NB), 128, 0, stream>>>(ATT, x, YP);
  gemm_pa(YP, P_outW, outB, T1, ROWS, SC, C_);
  k_ln_res_gelu<<<LNB, 256, 0, stream>>>(T1, x, Yb, ROWS);
  gemm(Yb, P_ffW, ffB, T1, nullptr, nullptr, ROWS, C_, C_, 0);
  k_ln_res_gelu<<<LNB, 256, 0, stream>>>(T1, x, Yb, ROWS);

  // ---- temporal blocks ----
  auto temporal = [&](const v16bf* PwIn, const float* bIn, const v16bf* PwOut,
                      const float* bOut, int upper, float* Zdst) {
    gemm(Yb, PwIn, bIn, nullptr, QR, KR, ROWS, C_, QKC, 2);
    k_qkT<<<dim3(1, NB), 128, 0, stream>>>(QR, KR, ATT, T_, V_ * IC_);
    k_att_norm_temporal<<<(N_ * S_ * T_ + 3) / 4, 128, 0, stream>>>(ATT, upper);
    k_temporal_apply<<<dim3(200, NB), 128, 0, stream>>>(ATT, Yb, YP);
    gemm_pa(YP, PwOut, bOut, T1, ROWS, SC, C_);
    k_ln_res_gelu<<<LNB, 256, 0, stream>>>(T1, nullptr, Zdst, ROWS);
  };
  temporal(P_tfW, tfB, P_ofW, ofB, /*tril*/ 0, ZF);
  temporal(P_tbW, tbB, P_obW, obB, /*triu*/ 1, ZB);

  // ---- final merge + ff ----
  long nel = (long)ROWS * C_;
  k_add3_gelu<<<(int)((nel + 255) / 256), 256, 0, stream>>>(Yb, ZF, ZB, Zb, nel);
  gemm(Zb, P_ftW, ftB, T1, nullptr, nullptr, ROWS, C_, C_, 0);
  k_ln_res_gelu<<<LNB, 256, 0, stream>>>(T1, Yb, out, ROWS);
}
